// AttentionLSTMDecoderV1_38869454029309
// MI455X (gfx1250) — compile-verified, matrix operand order checked by reading, not run on hardware
//
#include <hip/hip_runtime.h>
#include <hip/hip_bf16.h>

using bf16 = __hip_bfloat16;

typedef __attribute__((ext_vector_type(16))) __bf16 v16bf;
typedef __attribute__((ext_vector_type(8)))  float  v8f;
typedef __attribute__((ext_vector_type(4)))  int    i32x4;

union FragU { unsigned int u[8]; v16bf v; };

#if __has_builtin(__builtin_amdgcn_global_load_async_to_lds_b128) && \
    __has_builtin(__builtin_amdgcn_s_wait_asynccnt)
#define USE_ASYNC_LDS 1
typedef __attribute__((address_space(1))) i32x4* gptr128;  // global int4*
typedef __attribute__((address_space(3))) i32x4* lptr128;  // LDS int4*
__device__ inline void async_copy16(const void* g, void* l) {
  __builtin_amdgcn_global_load_async_to_lds_b128((gptr128)g, (lptr128)l, 0, 0);
}
#else
#define USE_ASYNC_LDS 0
#endif

// Load one 16x32 bf16 WMMA fragment (A or B operand) from a row-major
// [16 rows x K] tile. Per CDNA5 ISA 7.12.2: lane L holds row (L&15); half j
// maps to K = (j&7) + ((j>>3)<<4) + ((L>>4)<<3)  -> two contiguous 16B chunks
// at k-offsets {base, base+16} where base = (L>>4)*8.
__device__ inline v16bf load_bf16_frag(const bf16* p /* row ptr + koff */) {
  FragU f;
  *(uint4*)&f.u[0] = *(const uint4*)(p);
  *(uint4*)&f.u[4] = *(const uint4*)(p + 16);
  return f.v;
}

__device__ inline float sigmoidf_(float x) { return 1.0f / (1.0f + __expf(-x)); }

// ---------------------------------------------------------------------------
// Generic bf16 GEMM: C[M,N](f32 or bf16) = A[M,K](bf16) @ W[N,K]^T(bf16)
//                    (+ bias[n]) (+ Cin accumulate).  Requires K % 64 == 0.
// One wave per 16x64 output strip: grid = (N/64, M/16), block = 32.
// Ping-pong pipelined over two fragment sets: each set is reloaded in place
// only after its WMMAs consumed it (dead value -> no register copies).
// ---------------------------------------------------------------------------
#define GEMM_NT 4
__global__ __launch_bounds__(32) void gemm_bf16_wmma(
    const bf16* __restrict__ Ag, int lda,
    const bf16* __restrict__ Wg, int ldw,
    void* __restrict__ Cv, int ldc,
    const float* __restrict__ bias,
    const float* __restrict__ Cin,
    int K, int c_bf16)
{
  const int lane = threadIdx.x;
  const int row  = lane & 15;
  const int koff = (lane >> 4) << 3;          // 0 or 8
  const size_t m0 = (size_t)blockIdx.y * 16;
  const size_t n0 = (size_t)blockIdx.x * (16 * GEMM_NT);

  const bf16* aP = Ag + (m0 + row) * (size_t)lda + koff;
  const bf16* wP[GEMM_NT];
#pragma unroll
  for (int s = 0; s < GEMM_NT; ++s)
    wP[s] = Wg + (n0 + s * 16 + row) * (size_t)ldw + koff;

  const int mrow = (lane >> 4) * 8;           // C layout: M = r + (lane>=16)*8
  const int ncol = lane & 15;                 //            N = lane & 15

  v8f acc[GEMM_NT];
#pragma unroll
  for (int s = 0; s < GEMM_NT; ++s) {
    v8f a;
    if (Cin) {
#pragma unroll
      for (int r = 0; r < 8; ++r)
        a[r] = Cin[(m0 + mrow + r) * (size_t)ldc + n0 + s * 16 + ncol];
    } else {
#pragma unroll
      for (int r = 0; r < 8; ++r) a[r] = 0.0f;
    }
    acc[s] = a;
  }

  // set 0 prologue (K >= 64, K % 64 == 0)
  v16bf a0 = load_bf16_frag(aP);
  v16bf b0[GEMM_NT];
#pragma unroll
  for (int s = 0; s < GEMM_NT; ++s) b0[s] = load_bf16_frag(wP[s]);

  for (int k = 0; k < K; k += 64) {
    __builtin_prefetch(aP + k + 128, 0, 1);   // global_prefetch_b8
    // set 1 loads (k+32) in flight while set 0 WMMAs run
    v16bf a1 = load_bf16_frag(aP + k + 32);
    v16bf b1[GEMM_NT];
#pragma unroll
    for (int s = 0; s < GEMM_NT; ++s) b1[s] = load_bf16_frag(wP[s] + k + 32);
#pragma unroll
    for (int s = 0; s < GEMM_NT; ++s)
      acc[s] = __builtin_amdgcn_wmma_f32_16x16x32_bf16(
          false, a0, false, b0[s], (short)0, acc[s], false, false);
    if (k + 64 < K) {                         // set 0 reload (k+64), no copies
      a0 = load_bf16_frag(aP + k + 64);
#pragma unroll
      for (int s = 0; s < GEMM_NT; ++s) b0[s] = load_bf16_frag(wP[s] + k + 64);
    }
#pragma unroll
    for (int s = 0; s < GEMM_NT; ++s)
      acc[s] = __builtin_amdgcn_wmma_f32_16x16x32_bf16(
          false, a1, false, b1[s], (short)0, acc[s], false, false);
  }

#pragma unroll
  for (int s = 0; s < GEMM_NT; ++s) {
    const size_t n = n0 + s * 16 + ncol;
    const float bvl = bias ? bias[n] : 0.0f;
#pragma unroll
    for (int r = 0; r < 8; ++r) {
      const float val = acc[s][r] + bvl;
      const size_t idx = (m0 + mrow + r) * (size_t)ldc + n;
      if (c_bf16) ((bf16*)Cv)[idx] = __float2bfloat16(val);
      else        ((float*)Cv)[idx] = val;
    }
  }
}

// ---------------------------------------------------------------------------
__global__ void cvt_f32_to_bf16(const float* __restrict__ src,
                                bf16* __restrict__ dst, long n)
{
  long i = (long)blockIdx.x * blockDim.x + threadIdx.x;
  if (i < n) dst[i] = __float2bfloat16(src[i]);
}

// ---------------------------------------------------------------------------
// Shifted-label embedding gather -> bf16 into ro_in[..., H : H+E]
// grid = B*S blocks, block = E threads
// ---------------------------------------------------------------------------
__global__ void gather_embed(const float* __restrict__ embed,
                             const int* __restrict__ labels,
                             bf16* __restrict__ ro_in,
                             int S, int E, int LDRO, int OFF)
{
  const int bs = blockIdx.x;
  const int b = bs / S, s = bs % S;
  const int e = threadIdx.x;
  float val = 0.0f;
  if (s > 0) {
    int lab = labels[b * S + s - 1];
    val = embed[(size_t)lab * E + e];
  }
  ro_in[(size_t)bs * LDRO + OFF + e] = __float2bfloat16(val);
}

// ---------------------------------------------------------------------------
// inv_fert[b,t] = sigmoid(dot(enc[b,t,:], W_fert))   (one wave per (b,t))
// grid = (T/8, B), block = 256
// ---------------------------------------------------------------------------
__global__ __launch_bounds__(256) void fert_kernel(
    const float* __restrict__ enc, const float* __restrict__ W_fert,
    float* __restrict__ inv_fert, int T, int D)
{
  __shared__ float wf[512];
  const int b = blockIdx.y;
  for (int i = threadIdx.x; i < D; i += 256) wf[i] = W_fert[i];
  __syncthreads();
  const int wave = threadIdx.x >> 5, lane = threadIdx.x & 31;
  const int t = blockIdx.x * 8 + wave;
  const float* rowp = enc + ((size_t)b * T + t) * D;
  float s = 0.0f;
  for (int j = 0; j < 512 / 32; ++j) {
    int d = j * 32 + lane;
    s += rowp[d] * wf[d];
  }
  for (int off = 16; off > 0; off >>= 1) s += __shfl_down(s, off);
  if (lane == 0) inv_fert[b * T + t] = sigmoidf_(s);
}

// ---------------------------------------------------------------------------
// x = [emb_t (from ro_in), bf16(ctx)]   -> x_bf [B, E+D]
// ---------------------------------------------------------------------------
__global__ void concat_x_kernel(const bf16* __restrict__ ro_in,
                                const float* __restrict__ ctx,
                                bf16* __restrict__ x,
                                int step, int S, int E, int D, int LDRO, int HOFF)
{
  const int b = blockIdx.y;
  const int i = blockIdx.x * 256 + threadIdx.x;
  if (i >= E + D) return;
  bf16 v;
  if (i < E) v = ro_in[((size_t)b * S + step) * LDRO + HOFF + i];
  else       v = __float2bfloat16(ctx[(size_t)b * D + (i - E)]);
  x[(size_t)b * (E + D) + i] = v;
}

// ---------------------------------------------------------------------------
// LSTM cell + zoneout. grid = (H/256, B)
// ---------------------------------------------------------------------------
__global__ void lstm_kernel(const float* __restrict__ gates,
                            const float* __restrict__ b_ih,
                            const float* __restrict__ b_hh,
                            float* __restrict__ h, float* __restrict__ c,
                            bf16* __restrict__ h_bf, bf16* __restrict__ ro_in,
                            int step, int H, int S, int LDRO)
{
  const int b = blockIdx.y;
  const int j = blockIdx.x * 256 + threadIdx.x;
  const float* g = gates + (size_t)b * 4 * H;
  const float gi = g[j]         + b_ih[j]         + b_hh[j];
  const float gf = g[H + j]     + b_ih[H + j]     + b_hh[H + j];
  const float gg = g[2 * H + j] + b_ih[2 * H + j] + b_hh[2 * H + j];
  const float go = g[3 * H + j] + b_ih[3 * H + j] + b_hh[3 * H + j];
  const size_t idx = (size_t)b * H + j;
  const float c_old = c[idx], h_old = h[idx];
  const float c_new = sigmoidf_(gf) * c_old + sigmoidf_(gi) * tanhf(gg);
  const float h_new = sigmoidf_(go) * tanhf(c_new);
  const float hz = 0.05f * h_old + 0.95f * h_new;
  const float cz = 0.15f * c_old + 0.85f * c_new;
  h[idx] = hz;
  c[idx] = cz;
  const bf16 hb = __float2bfloat16(hz);
  h_bf[idx] = hb;
  ro_in[((size_t)b * S + step) * LDRO + j] = hb;
}

// ---------------------------------------------------------------------------
// energies[b,t] = sum_a tanh(enc_ctx[b,t,a] + s_t[b,a] + accum[b,t]*W_fb[a]) * v_att[a]
// One wave handles ENER_CH consecutive t's; next row staged into LDS via
// async global->LDS copies (when available) while the current row computes.
// grid = (T/(8*ENER_CH), B), block = 256 (8 waves).
// ---------------------------------------------------------------------------
#define ENER_CH 5
__global__ __launch_bounds__(256) void energies_kernel(
    const bf16* __restrict__ enc_ctx, const float* __restrict__ s_t,
    const float* __restrict__ accum, const float* __restrict__ W_fb,
    const float* __restrict__ v_att, const int* __restrict__ enc_len,
    float* __restrict__ energies, int T, int A)
{
  __shared__ float sh_s[1024];
  __shared__ float sh_fb[1024];
  __shared__ float sh_v[1024];
#if USE_ASYNC_LDS
  __shared__ __align__(16) unsigned int stage[8][2][512];  // 8 waves x dbl x 2KB
#endif
  const int b = blockIdx.y;
  for (int i = threadIdx.x; i < A; i += 256) {
    sh_s[i]  = s_t[(size_t)b * A + i];
    sh_fb[i] = W_fb[i];
    sh_v[i]  = v_att[i];
  }
  __syncthreads();
  const int wave = threadIdx.x >> 5, lane = threadIdx.x & 31;
  const int tbase = (blockIdx.x * 8 + wave) * ENER_CH;
  const int elen = enc_len[b];
  const int abase = lane * 32;

#if USE_ASYNC_LDS
  {
    const char* g = (const char*)(enc_ctx + ((size_t)b * T + tbase) * A) + lane * 16;
    char* l = (char*)&stage[wave][0][0] + lane * 16;
#pragma unroll
    for (int cc = 0; cc < 4; ++cc) async_copy16(g + cc * 512, l + cc * 512);
  }
#endif
  for (int i = 0; i < ENER_CH; ++i) {
    const int t = tbase + i;
    const float ac = accum[(size_t)b * T + t];
#if USE_ASYNC_LDS
    if (i + 1 < ENER_CH) {
      const char* g = (const char*)(enc_ctx + ((size_t)b * T + t + 1) * A) + lane * 16;
      char* l = (char*)&stage[wave][(i + 1) & 1][0] + lane * 16;
#if __has_builtin(__builtin_amdgcn_s_wait_dscnt)
      __builtin_amdgcn_s_wait_dscnt(0);   // prior reads of this buffer done
#endif
#pragma unroll
      for (int cc = 0; cc < 4; ++cc) async_copy16(g + cc * 512, l + cc * 512);
      __builtin_amdgcn_s_wait_asynccnt(4);  // current row landed; next in flight
    } else {
      __builtin_amdgcn_s_wait_asynccnt(0);
    }
    const unsigned int* rowp = &stage[wave][i & 1][0] + lane * 16;
#else
    const unsigned int* rowp =
        (const unsigned int*)(enc_ctx + ((size_t)b * T + t) * A) + lane * 16;
#endif
    float sum = 0.0f;
#pragma unroll
    for (int cidx = 0; cidx < 16; ++cidx) {
      const unsigned int u = rowp[cidx];
      const float e0 = __uint_as_float(u << 16);
      const float e1 = __uint_as_float(u & 0xffff0000u);
      const int a = abase + 2 * cidx;
      sum += tanhf(e0 + sh_s[a]     + ac * sh_fb[a])     * sh_v[a];
      sum += tanhf(e1 + sh_s[a + 1] + ac * sh_fb[a + 1]) * sh_v[a + 1];
    }
    for (int off = 16; off > 0; off >>= 1) sum += __shfl_down(sum, off);
    if (lane == 0)
      energies[(size_t)b * T + t] = (t < elen) ? sum : -1e30f;
  }
}

// ---------------------------------------------------------------------------
// Masked softmax over T + coverage accum update. grid = B, block 256.
// ---------------------------------------------------------------------------
__global__ __launch_bounds__(256) void softmax_accum_kernel(
    const float* __restrict__ energies, const float* __restrict__ inv_fert,
    float* __restrict__ w_att, float* __restrict__ accum, int T)
{
  const int b = blockIdx.x, tid = threadIdx.x;
  __shared__ float red[256];
  float m = -3.4e38f;
  for (int t = tid; t < T; t += 256) m = fmaxf(m, energies[(size_t)b * T + t]);
  red[tid] = m;
  __syncthreads();
  for (int s = 128; s > 0; s >>= 1) {
    if (tid < s) red[tid] = fmaxf(red[tid], red[tid + s]);
    __syncthreads();
  }
  m = red[0];
  __syncthreads();
  float sum = 0.0f;
  for (int t = tid; t < T; t += 256) sum += __expf(energies[(size_t)b * T + t] - m);
  red[tid] = sum;
  __syncthreads();
  for (int s = 128; s > 0; s >>= 1) {
    if (tid < s) red[tid] += red[tid + s];
    __syncthreads();
  }
  const float inv = 1.0f / red[0];
  for (int t = tid; t < T; t += 256) {
    const size_t i = (size_t)b * T + t;
    const float w = __expf(energies[i] - m) * inv;
    w_att[i] = w;
    accum[i] += w * inv_fert[i] * 0.5f;
  }
}

// ---------------------------------------------------------------------------
// ctx partial sums over T-slices: part[b,slice,d] = sum_{t in slice} w*enc
// grid = (NS, B), block = D/2 (uint-packed bf16 pairs)
// ---------------------------------------------------------------------------
__global__ __launch_bounds__(256) void ctx_partial_kernel(
    const bf16* __restrict__ enc_bf, const float* __restrict__ w_att,
    float* __restrict__ part, int T, int D, int NS)
{
  const int b = blockIdx.y, slice = blockIdx.x, d2 = threadIdx.x;
  const int tlen = T / NS;
  const int t0 = slice * tlen;
  const unsigned int* p =
      (const unsigned int*)(enc_bf + (size_t)b * T * D) + (size_t)t0 * (D / 2) + d2;
  const float* w = w_att + (size_t)b * T + t0;
  float s0 = 0.0f, s1 = 0.0f;
  for (int t = 0; t < tlen; ++t) {
    const unsigned int u = p[(size_t)t * (D / 2)];
    const float wv = w[t];
    s0 += wv * __uint_as_float(u << 16);
    s1 += wv * __uint_as_float(u & 0xffff0000u);
  }
  float* dst = part + ((size_t)b * NS + slice) * D + 2 * d2;
  dst[0] = s0;
  dst[1] = s1;
}

// grid = B, block = D
__global__ void ctx_reduce_kernel(const float* __restrict__ part,
                                  float* __restrict__ ctx, bf16* __restrict__ ro_in,
                                  int step, int D, int S, int LDRO, int OFF, int NS)
{
  const int b = blockIdx.x, d = threadIdx.x;
  float s = 0.0f;
  for (int k = 0; k < NS; ++k) s += part[((size_t)b * NS + k) * D + d];
  ctx[(size_t)b * D + d] = s;
  ro_in[((size_t)b * S + step) * LDRO + OFF + d] = __float2bfloat16(s);
}

// ---------------------------------------------------------------------------
// MaxOut over adjacent pairs -> bf16. n = M * (P/2)
// ---------------------------------------------------------------------------
__global__ void maxout_kernel(const float* __restrict__ ro, bf16* __restrict__ out,
                              long n, int Ph)
{
  long i = (long)blockIdx.x * 256 + threadIdx.x;
  if (i >= n) return;
  const long m = i / Ph;
  const int p = (int)(i % Ph);
  const float v = fmaxf(ro[m * 2 * Ph + 2 * p], ro[m * 2 * Ph + 2 * p + 1]);
  out[i] = __float2bfloat16(v);
}

// ===========================================================================
extern "C" void kernel_launch(void* const* d_in, const int* in_sizes, int n_in,
                              void* d_out, int out_size, void* d_ws, size_t ws_size,
                              hipStream_t stream)
{
  (void)in_sizes; (void)n_in; (void)out_size; (void)ws_size;
  constexpr int B = 32, T = 1000, S = 100, D = 512, A = 1024, H = 1024,
                E = 640, V = 10240, P = 1024;
  constexpr int G4H = 4 * H, KX = E + D /*1152*/, KRO = H + E + D /*2176*/;
  constexpr int NS = 8;  // context T-slices

  const float* enc    = (const float*)d_in[0];
  const int*   labels = (const int*)  d_in[1];
  const int*   elen   = (const int*)  d_in[2];
  const float* embed  = (const float*)d_in[3];
  const float* W_ih   = (const float*)d_in[4];
  const float* W_hh   = (const float*)d_in[5];
  const float* b_ih   = (const float*)d_in[6];
  const float* b_hh   = (const float*)d_in[7];
  const float* W_s    = (const float*)d_in[8];
  const float* W_enc  = (const float*)d_in[9];
  const float* b_enc  = (const float*)d_in[10];
  const float* v_att  = (const float*)d_in[11];
  const float* W_fert = (const float*)d_in[12];
  const float* W_fb   = (const float*)d_in[13];
  const float* W_ro   = (const float*)d_in[14];
  const float* b_ro   = (const float*)d_in[15];
  const float* W_out  = (const float*)d_in[16];
  const float* b_out  = (const float*)d_in[17];
  float* logits = (float*)d_out;

  char* wsp = (char*)d_ws;
  auto carve = [&](size_t bytes) -> void* {
    void* p = wsp;
    wsp += (bytes + 255) & ~(size_t)255;
    return p;
  };

  bf16* enc_bf  = (bf16*)carve((size_t)B * T * D * 2);
  bf16* ectx_bf = (bf16*)carve((size_t)B * T * A * 2);
  bf16* ro_in   = (bf16*)carve((size_t)B * S * KRO * 2);
  bf16* Wih_bf  = (bf16*)carve((size_t)G4H * KX * 2);
  bf16* Whh_bf  = (bf16*)carve((size_t)G4H * H * 2);
  bf16* Ws_bf   = (bf16*)carve((size_t)A * H * 2);
  bf16* Wenc_bf = (bf16*)carve((size_t)A * D * 2);
  bf16* Wro_bf  = (bf16*)carve((size_t)P * KRO * 2);
  bf16* Wout_bf = (bf16*)carve((size_t)V * (P / 2) * 2);
  float* gates  = (float*)carve((size_t)B * G4H * 4);
  float* s_t    = (float*)carve((size_t)B * A * 4);
  float* hst    = (float*)carve((size_t)B * H * 4);
  float* cst    = (float*)carve((size_t)B * H * 4);
  float* ctx    = (float*)carve((size_t)B * D * 4);
  bf16* h_bf    = (bf16*)carve((size_t)B * H * 2);
  bf16* x_bf    = (bf16*)carve((size_t)B * KX * 2);
  float* accum  = (float*)carve((size_t)B * T * 4);
  float* ener   = (float*)carve((size_t)B * T * 4);
  float* watt   = (float*)carve((size_t)B * T * 4);
  float* ifert  = (float*)carve((size_t)B * T * 4);
  float* ctxp   = (float*)carve((size_t)B * NS * D * 4);
  float* ro_f   = (float*)carve((size_t)B * S * P * 4);
  bf16* ro_mx   = (bf16*)carve((size_t)B * S * (P / 2) * 2);

  (void)hipMemsetAsync(hst,   0, (size_t)B * H * 4, stream);
  (void)hipMemsetAsync(cst,   0, (size_t)B * H * 4, stream);
  (void)hipMemsetAsync(ctx,   0, (size_t)B * D * 4, stream);
  (void)hipMemsetAsync(h_bf,  0, (size_t)B * H * 2, stream);
  (void)hipMemsetAsync(accum, 0, (size_t)B * T * 4, stream);

  auto cvt = [&](const float* s, bf16* dst, long n) {
    cvt_f32_to_bf16<<<dim3((unsigned)((n + 255) / 256)), dim3(256), 0, stream>>>(s, dst, n);
  };
  cvt(enc,   enc_bf,  (long)B * T * D);
  cvt(W_ih,  Wih_bf,  (long)G4H * KX);
  cvt(W_hh,  Whh_bf,  (long)G4H * H);
  cvt(W_s,   Ws_bf,   (long)A * H);
  cvt(W_enc, Wenc_bf, (long)A * D);
  cvt(W_ro,  Wro_bf,  (long)P * KRO);
  cvt(W_out, Wout_bf, (long)V * (P / 2));

  gather_embed<<<dim3(B * S), dim3(E), 0, stream>>>(embed, labels, ro_in, S, E, KRO, H);

  // enc_ctx = enc @ W_enc^T + b_enc -> bf16 [B*T, A]
  gemm_bf16_wmma<<<dim3(A / 64, (B * T) / 16), dim3(32), 0, stream>>>(
      enc_bf, D, Wenc_bf, D, ectx_bf, A, b_enc, nullptr, D, 1);

  fert_kernel<<<dim3(T / 8, B), dim3(256), 0, stream>>>(enc, W_fert, ifert, T, D);

  for (int t = 0; t < S; ++t) {
    concat_x_kernel<<<dim3((KX + 255) / 256, B), dim3(256), 0, stream>>>(
        ro_in, ctx, x_bf, t, S, E, D, KRO, H);
    gemm_bf16_wmma<<<dim3(G4H / 64, B / 16), dim3(32), 0, stream>>>(
        x_bf, KX, Wih_bf, KX, gates, G4H, nullptr, nullptr, KX, 0);
    gemm_bf16_wmma<<<dim3(G4H / 64, B / 16), dim3(32), 0, stream>>>(
        h_bf, H, Whh_bf, H, gates, G4H, nullptr, gates, H, 0);
    lstm_kernel<<<dim3(H / 256, B), dim3(256), 0, stream>>>(
        gates, b_ih, b_hh, hst, cst, h_bf, ro_in, t, H, S, KRO);
    gemm_bf16_wmma<<<dim3(A / 64, B / 16), dim3(32), 0, stream>>>(
        h_bf, H, Ws_bf, H, s_t, A, nullptr, nullptr, H, 0);
    energies_kernel<<<dim3(T / (8 * ENER_CH), B), dim3(256), 0, stream>>>(
        ectx_bf, s_t, accum, W_fb, v_att, elen, ener, T, A);
    softmax_accum_kernel<<<dim3(B), dim3(256), 0, stream>>>(ener, ifert, watt, accum, T);
    ctx_partial_kernel<<<dim3(NS, B), dim3(D / 2), 0, stream>>>(
        enc_bf, watt, ctxp, T, D, NS);
    ctx_reduce_kernel<<<dim3(B), dim3(D), 0, stream>>>(
        ctxp, ctx, ro_in, t, D, S, KRO, H + E, NS);
  }

  gemm_bf16_wmma<<<dim3(P / 64, (B * S) / 16), dim3(32), 0, stream>>>(
      ro_in, KRO, Wro_bf, KRO, ro_f, P, b_ro, nullptr, KRO, 0);
  maxout_kernel<<<dim3((unsigned)(((long)B * S * (P / 2) + 255) / 256)), dim3(256), 0, stream>>>(
      ro_f, ro_mx, (long)B * S * (P / 2), P / 2);
  gemm_bf16_wmma<<<dim3(V / 64, (B * S) / 16), dim3(32), 0, stream>>>(
      ro_mx, P / 2, Wout_bf, P / 2, logits, V, b_out, nullptr, P / 2, 0);
}
